// SelfAttention_25074019074592
// MI455X (gfx1250) — matrix operand order repaired and checked by reading, not compiled, not run on hardware
//
#include <hip/hip_runtime.h>
#include <math.h>

// ---------------- problem constants ----------------
static constexpr int cB = 4;
static constexpr int cL = 4096;
static constexpr int cE = 512;
static constexpr int cH = 8;
static constexpr int cD = 64;    // head dim
static constexpr int cM = 256;   // random features
static constexpr int cNCH = 16;  // chunks of 256 rows

// ---------------- numeric constants ----------------
#define DN_SCALE 0.35355339059327373f   // 64^-0.25
#define RATIO    0.0625f                // 256^-0.5
#define FEPS     1e-4f

// ---------------- workspace layout (bytes) ----------------
static constexpr size_t OFF_QPHI = 0;                                    // bf16 [B,H,L,M]
static constexpr size_t SZ_PHI   = (size_t)cB * cH * cL * cM * 2;        // 64 MB
static constexpr size_t OFF_KPHI = OFF_QPHI + SZ_PHI;
static constexpr size_t OFF_VB   = OFF_KPHI + SZ_PHI;                    // bf16 [B,H,L,D]
static constexpr size_t SZ_VB    = (size_t)cB * cH * cL * cD * 2;
static constexpr size_t OFF_KH   = OFF_VB + SZ_VB;                       // f32 [B,H,L,D]
static constexpr size_t SZ_KH    = (size_t)cB * cH * cL * cD * 4;
static constexpr size_t OFF_KD   = OFF_KH + SZ_KH;                       // f32 [B,H,L]
static constexpr size_t SZ_KD    = (size_t)cB * cH * cL * 4;
static constexpr size_t OFF_ATTN = OFF_KD + SZ_KD;                       // bf16 [B,L,E]
static constexpr size_t SZ_ATTN  = (size_t)cB * cL * cE * 2;
static constexpr size_t OFF_GMAX = OFF_ATTN + SZ_ATTN;                   // int (ordered float)
static constexpr size_t OFF_SC   = OFF_GMAX + 256;                       // f32 [B*H,NCH,M,D]
static constexpr size_t SZ_SC    = (size_t)cB * cH * cNCH * cM * cD * 4; // 32 MB
static constexpr size_t OFF_ZC   = OFF_SC + SZ_SC;                       // f32 [B*H,NCH,M]

// ---------------- WMMA helpers (gfx1250, wave32) ----------------
typedef __bf16 bf16x16 __attribute__((ext_vector_type(16)));
typedef float  f32x8   __attribute__((ext_vector_type(8)));

__device__ __forceinline__ f32x8 zero8() {
  f32x8 z = {0.f, 0.f, 0.f, 0.f, 0.f, 0.f, 0.f, 0.f};
  return z;
}

__device__ __forceinline__ f32x8 wmma_bf(bf16x16 a, bf16x16 b, f32x8 c) {
  return __builtin_amdgcn_wmma_f32_16x16x32_bf16(false, a, false, b, (short)0, c, false, false);
}

// A fragment (16x32 bf16): element(m,k) at p[m*rs + k*cs]
__device__ __forceinline__ bf16x16 frag_a_bf(const __bf16* p, int rs, int cs, int lane) {
  bf16x16 f;
  const int m  = lane & 15;
  const int kb = (lane & 16) ? 16 : 0;
#pragma unroll
  for (int i = 0; i < 16; ++i) {
    const int k = kb + i;  // contiguous: lanes 0-15 -> K 0..15, lanes 16-31 -> K 16..31
    f[i] = p[m * rs + k * cs];
  }
  return f;
}

// B fragment (32x16 bf16): element(k,n) at p[k*rs + n*cs]
__device__ __forceinline__ bf16x16 frag_b_bf(const __bf16* p, int rs, int cs, int lane) {
  bf16x16 f;
  const int n  = lane & 15;
  const int kb = (lane & 16) ? 16 : 0;
#pragma unroll
  for (int i = 0; i < 16; ++i) f[i] = p[(kb + i) * rs + n * cs];
  return f;
}

// B fragment sourced from f32 (converted on the fly)
__device__ __forceinline__ bf16x16 frag_b_f32(const float* p, int rs, int cs, int lane) {
  bf16x16 f;
  const int n  = lane & 15;
  const int kb = (lane & 16) ? 16 : 0;
#pragma unroll
  for (int i = 0; i < 16; ++i) f[i] = (__bf16)p[(kb + i) * rs + n * cs];
  return f;
}

// C/D fragment: element(m,n) at p[m*rs + n*cs]; 8 VGPRs, M = r + 8*(lane>=16)
__device__ __forceinline__ void stC(float* p, int rs, int cs, int lane, f32x8 c) {
  const int n = lane & 15;
  const int mb = (lane & 16) ? 8 : 0;
#pragma unroll
  for (int r = 0; r < 8; ++r) p[(mb + r) * rs + n * cs] = c[r];
}
__device__ __forceinline__ f32x8 ldC(const float* p, int rs, int cs, int lane) {
  const int n = lane & 15;
  const int mb = (lane & 16) ? 8 : 0;
  f32x8 c;
#pragma unroll
  for (int r = 0; r < 8; ++r) c[r] = p[(mb + r) * rs + n * cs];
  return c;
}

// ordered-int float max encoding
__device__ __forceinline__ int f2oi(float f) {
  int i = __float_as_int(f);
  return i >= 0 ? i : (i ^ 0x7FFFFFFF);
}
__device__ __forceinline__ float oi2f(int i) {
  return __int_as_float(i >= 0 ? i : (i ^ 0x7FFFFFFF));
}

// ---------------- shared staging / GEMM building blocks ----------------
__device__ __forceinline__ void stage_x64(const float* g, __bf16* s, int tid) {
  for (int i = tid; i < 64 * 64; i += 256)
    s[i] = (__bf16)g[(size_t)(i >> 6) * cE + (i & 63)];
}
__device__ __forceinline__ void stage_w64(const float* g, __bf16* s, int tid) {
  for (int i = tid; i < 64 * 64; i += 256) s[i] = (__bf16)g[i];
}

// D[64,64] = A[64,64] @ W^T   (W stored row-major [j][d]; B(k=d,n=j)=W[j*64+d])
__device__ __forceinline__ void gemm_head(const __bf16* sA, const __bf16* sW, float* sD,
                                          int wave, int lane) {
  for (int t = wave; t < 16; t += 8) {
    const int mt = (t >> 2) << 4, nt = (t & 3) << 4;
    f32x8 acc = zero8();
#pragma unroll
    for (int kk = 0; kk < 64; kk += 32) {
      bf16x16 a = frag_a_bf(sA + mt * 64 + kk, 64, 1, lane);
      bf16x16 b = frag_b_bf(sW + nt * 64 + kk, 1, 64, lane);
      acc = wmma_bf(a, b, acc);
    }
    stC(sD + mt * 64 + nt, 64, 1, lane, acc);
  }
}

// Xd[64,256] = A[64,64] @ proj^T  (proj row-major [m][d]; B(k=d,n=m)=proj[m*64+d])
__device__ __forceinline__ void gemm_xd(const __bf16* sA, const __bf16* sProj, float* sXd,
                                        int wave, int lane) {
  for (int t = wave; t < 64; t += 8) {
    const int mt = (t >> 4) << 4, nt = (t & 15) << 4;
    f32x8 acc = zero8();
#pragma unroll
    for (int kk = 0; kk < 64; kk += 32) {
      bf16x16 a = frag_a_bf(sA + mt * 64 + kk, 64, 1, lane);
      bf16x16 b = frag_b_bf(sProj + nt * 64 + kk, 1, 64, lane);
      acc = wmma_bf(a, b, acc);
    }
    stC(sXd + mt * cM + nt, cM, 1, lane, acc);
  }
}

// ---------------- kernel 0: init global max ----------------
__global__ void k0_init(int* gmax) {
  if (threadIdx.x == 0) *gmax = (int)0x80000000;
}

// ---------------- kernel 1: head projections + q-phi + k stats ----------------
__global__ void __launch_bounds__(256)
k1_qkv_phi(const float* __restrict__ qx, const float* __restrict__ kx,
           const float* __restrict__ vx, const float* __restrict__ Wq,
           const float* __restrict__ Wk, const float* __restrict__ Wv,
           const float* __restrict__ proj,
           __bf16* __restrict__ qphi, float* __restrict__ khb,
           __bf16* __restrict__ vb, float* __restrict__ kdiag, int* gmax) {
  const int tid = threadIdx.x, wave = tid >> 5, lane = tid & 31;
  const int bh = blockIdx.y, b = bh >> 3, h = bh & 7;
  const int l0 = blockIdx.x * 64;

  __shared__ __bf16 sProj[cM * cD];   // 32 KB
  __shared__ __bf16 sIn[64 * 64];     // 8 KB
  __shared__ __bf16 sW[64 * 64];      // 8 KB
  __shared__ float  sHead[64 * 64];   // 16 KB
  __shared__ __bf16 sHB[64 * 64];     // 8 KB
  __shared__ float  sXd[64 * cM];     // 64 KB
  __shared__ float  sRed[64];

  for (int i = tid; i < cM * cD; i += 256) sProj[i] = (__bf16)proj[i];

  const size_t inBase = ((size_t)b * cL + l0) * cE + (size_t)h * cD;
  const size_t hBase  = (size_t)bh * cL + l0;

  // ---- V path ----
  stage_x64(vx + inBase, sIn, tid);
  stage_w64(Wv, sW, tid);
  __syncthreads();
  gemm_head(sIn, sW, sHead, wave, lane);
  __syncthreads();
  for (int i = tid; i < 64 * 64; i += 256) vb[hBase * cD + i] = (__bf16)sHead[i];

  // ---- K path ----
  stage_x64(kx + inBase, sIn, tid);
  stage_w64(Wk, sW, tid);
  __syncthreads();
  gemm_head(sIn, sW, sHead, wave, lane);
  __syncthreads();
  for (int i = tid; i < 64 * 64; i += 256) khb[hBase * cD + i] = sHead[i];
  if (tid < 64) {
    float s = 0.f;
    for (int c = 0; c < 64; ++c) { float t = sHead[tid * 64 + c]; s += t * t; }
    kdiag[hBase + tid] = 0.5f * DN_SCALE * DN_SCALE * s;
  }
  for (int i = tid; i < 64 * 64; i += 256) sHB[i] = (__bf16)(sHead[i] * DN_SCALE);
  __syncthreads();
  gemm_xd(sHB, sProj, sXd, wave, lane);
  __syncthreads();
  {  // block max of k-features -> global atomic max
    float mx = -3.4e38f;
    for (int i = tid; i < 64 * cM; i += 256) mx = fmaxf(mx, sXd[i]);
#pragma unroll
    for (int off = 16; off > 0; off >>= 1) mx = fmaxf(mx, __shfl_xor(mx, off, 32));
    if (lane == 0) sRed[wave] = mx;
    __syncthreads();
    if (tid == 0) {
      float m2 = sRed[0];
      for (int w2 = 1; w2 < 8; ++w2) m2 = fmaxf(m2, sRed[w2]);
      atomicMax(gmax, f2oi(m2));
    }
    __syncthreads();
  }

  // ---- Q path ----
  stage_x64(qx + inBase, sIn, tid);
  stage_w64(Wq, sW, tid);
  __syncthreads();
  gemm_head(sIn, sW, sHead, wave, lane);
  __syncthreads();
  for (int i = tid; i < 64 * 64; i += 256) sHB[i] = (__bf16)(sHead[i] * DN_SCALE);
  __syncthreads();
  gemm_xd(sHB, sProj, sXd, wave, lane);
  __syncthreads();
  if (tid < 64) {  // rowmax + diag
    float mxr = -3.4e38f;
    for (int m = 0; m < cM; ++m) mxr = fmaxf(mxr, sXd[tid * cM + m]);
    float s = 0.f;
    for (int c = 0; c < 64; ++c) { float t = sHead[tid * 64 + c]; s += t * t; }
    sRed[tid] = mxr + 0.5f * DN_SCALE * DN_SCALE * s;
  }
  __syncthreads();
  for (int i = tid; i < 64 * cM; i += 256) {
    const int r = i >> 8;
    qphi[(hBase + r) * cM + (i & 255)] =
        (__bf16)(RATIO * (expf(sXd[i] - sRed[r]) + FEPS));
  }
}

// ---------------- kernel 2: finish k-phi with global max ----------------
__global__ void __launch_bounds__(256)
k2_kphi(const float* __restrict__ khb, const float* __restrict__ kdiag,
        const float* __restrict__ proj, const int* __restrict__ gmax,
        __bf16* __restrict__ kphi) {
  const int tid = threadIdx.x, wave = tid >> 5, lane = tid & 31;
  const int bh = blockIdx.y;
  const int l0 = blockIdx.x * 64;

  __shared__ __bf16 sProj[cM * cD];
  __shared__ __bf16 sHB[64 * 64];
  __shared__ float  sXd[64 * cM];

  for (int i = tid; i < cM * cD; i += 256) sProj[i] = (__bf16)proj[i];
  const size_t hBase = (size_t)bh * cL + l0;
  for (int i = tid; i < 64 * 64; i += 256)
    sHB[i] = (__bf16)(khb[hBase * cD + i] * DN_SCALE);
  __syncthreads();
  gemm_xd(sHB, sProj, sXd, wave, lane);
  __syncthreads();
  const float gm = oi2f(*gmax);
  for (int i = tid; i < 64 * cM; i += 256) {
    const int r = i >> 8;
    kphi[(hBase + r) * cM + (i & 255)] =
        (__bf16)(RATIO * (expf(sXd[i] - kdiag[hBase + r] - gm) + FEPS));
  }
}

// ---------------- kernel 3a: per-chunk state  S_c = k^T v, z_c = sum k ----------------
// grid (NCH, B*H); each block owns one 256-row chunk, accumulating over four
// 64-row stages with WMMA accumulators held in registers.
__global__ void __launch_bounds__(256)
k3a_chunk(const __bf16* __restrict__ kphi, const __bf16* __restrict__ vb,
          float* __restrict__ Sc, float* __restrict__ zc) {
  const int tid = threadIdx.x, wave = tid >> 5, lane = tid & 31;
  const int chunk = blockIdx.x, bh = blockIdx.y;

  __shared__ __bf16 sK[64 * cM];   // 32 KB
  __shared__ __bf16 sV[64 * cD];   // 8 KB

  f32x8 acc[8];
#pragma unroll
  for (int i = 0; i < 8; ++i) acc[i] = zero8();
  float zacc = 0.f;

  for (int sc = 0; sc < 4; ++sc) {
    const int l0 = chunk * 256 + sc * 64;
    const size_t rowPhi = ((size_t)bh * cL + l0) * cM;
    const size_t rowV   = ((size_t)bh * cL + l0) * cD;
    for (int i = tid; i < 64 * cM; i += 256) sK[i] = kphi[rowPhi + i];
    for (int i = tid; i < 64 * cD; i += 256) sV[i] = vb[rowV + i];
    __syncthreads();
#pragma unroll
    for (int i = 0; i < 8; ++i) {
      const int t = wave + i * 8;              // 64 tiles over S[256,64]
      const int mt = (t >> 2) << 4, nt = (t & 3) << 4;
#pragma unroll
      for (int kk = 0; kk < 64; kk += 32)
        acc[i] = wmma_bf(frag_a_bf(sK + kk * cM + mt, 1, cM, lane),
                         frag_b_bf(sV + kk * cD + nt, cD, 1, lane), acc[i]);
    }
    {
      float s = 0.f;
      for (int j = 0; j < 64; ++j) s += (float)sK[j * cM + tid];
      zacc += s;
    }
    __syncthreads();
  }

  const size_t scBase = ((size_t)bh * cNCH + chunk) * (cM * cD);
#pragma unroll
  for (int i = 0; i < 8; ++i) {
    const int t = wave + i * 8;
    const int mt = (t >> 2) << 4, nt = (t & 3) << 4;
    stC(Sc + scBase + mt * cD + nt, cD, 1, lane, acc[i]);
  }
  zc[((size_t)bh * cNCH + chunk) * cM + tid] = zacc;
}

// ---------------- kernel 3b: exclusive prefix over chunk states ----------------
__global__ void __launch_bounds__(256)
k3b_prefix(float* __restrict__ Sc, float* __restrict__ zc) {
  const int tid = threadIdx.x;
  const int bh = blockIdx.x;
  const size_t base = (size_t)bh * cNCH * cM * cD;
  for (int e = 0; e < cM * cD / 256; ++e) {
    const int idx = tid + e * 256;
    float run = 0.f;
    for (int c = 0; c < cNCH; ++c) {
      const size_t p = base + (size_t)c * cM * cD + idx;
      const float cur = Sc[p];
      Sc[p] = run;
      run += cur;
    }
  }
  const size_t zbase = (size_t)bh * cNCH * cM;
  float run = 0.f;
  for (int c = 0; c < cNCH; ++c) {
    const size_t p = zbase + (size_t)c * cM + tid;
    const float cur = zc[p];
    zc[p] = run;
    run += cur;
  }
}

// ---------------- kernel 3c: per-chunk causal output ----------------
// grid (NCH, B*H): start from exclusive-prefix state, run 4 intra-chunk
// 64-row steps (masked scores + q@S + scores@v, then local S,z update).
__global__ void __launch_bounds__(256)
k3c_out(const __bf16* __restrict__ qphi, const __bf16* __restrict__ kphi,
        const __bf16* __restrict__ vb, const float* __restrict__ Sc,
        const float* __restrict__ zc, __bf16* __restrict__ attnb) {
  const int tid = threadIdx.x, wave = tid >> 5, lane = tid & 31;
  const int chunk = blockIdx.x, bh = blockIdx.y, b = bh >> 3, h = bh & 7;

  __shared__ float  S[cM * cD];     // 64 KB running state
  __shared__ float  zv[cM];
  __shared__ __bf16 sQ[64 * cM];    // 32 KB
  __shared__ __bf16 sK[64 * cM];    // 32 KB
  __shared__ __bf16 sV[64 * cD];    // 8 KB
  __shared__ __bf16 sSc[64 * 64];   // 8 KB masked scores
  __shared__ float  sDen[64];

  const size_t scBase = ((size_t)bh * cNCH + chunk) * (cM * cD);
  for (int i = tid; i < cM * cD; i += 256) S[i] = Sc[scBase + i];
  zv[tid] = zc[((size_t)bh * cNCH + chunk) * cM + tid];
  __syncthreads();

  for (int sc = 0; sc < 4; ++sc) {
    const int l0 = chunk * 256 + sc * 64;
    const size_t rowPhi = ((size_t)bh * cL + l0) * cM;
    const size_t rowV   = ((size_t)bh * cL + l0) * cD;
    for (int i = tid; i < 64 * cM; i += 256) {
      sQ[i] = qphi[rowPhi + i];
      sK[i] = kphi[rowPhi + i];
    }
    for (int i = tid; i < 64 * cD; i += 256) sV[i] = vb[rowV + i];
    __syncthreads();

    // den base: q . z (state before this sub-chunk)
    if (tid < 64) {
      float s = 0.f;
      for (int m = 0; m < cM; ++m) s += (float)sQ[tid * cM + m] * zv[m];
      sDen[tid] = s;
    }
    __syncthreads();

    // scores = (q @ k^T) tril-masked; accumulate row sums into den
    for (int t = wave; t < 16; t += 8) {
      const int mt = (t >> 2) << 4, nt = (t & 3) << 4;
      f32x8 acc = zero8();
#pragma unroll
      for (int kk = 0; kk < cM; kk += 32) {
        bf16x16 a = frag_a_bf(sQ + mt * cM + kk, cM, 1, lane);
        bf16x16 bfr = frag_b_bf(sK + nt * cM + kk, 1, cM, lane);  // B(k=m,n=j)=sK[j*cM+m]
        acc = wmma_bf(a, bfr, acc);
      }
      const int n = lane & 15, mb = (lane & 16) ? 8 : 0;
#pragma unroll
      for (int r = 0; r < 8; ++r) {
        const int li = mt + mb + r, lj = nt + n;
        const float sv = (lj <= li) ? acc[r] : 0.f;
        sSc[li * 64 + lj] = (__bf16)sv;
        if (sv != 0.f) atomicAdd(&sDen[li], sv);
      }
    }
    __syncthreads();

    // num = scores @ v + q @ S ; out = num / den
    for (int t = wave; t < 16; t += 8) {
      const int mt = (t >> 2) << 4, nt = (t & 3) << 4;
      f32x8 acc = zero8();
#pragma unroll
      for (int kk = 0; kk < 64; kk += 32)
        acc = wmma_bf(frag_a_bf(sSc + mt * 64 + kk, 64, 1, lane),
                      frag_b_bf(sV + kk * cD + nt, cD, 1, lane), acc);
#pragma unroll
      for (int kk = 0; kk < cM; kk += 32)
        acc = wmma_bf(frag_a_bf(sQ + mt * cM + kk, cM, 1, lane),
                      frag_b_f32(S + kk * cD + nt, cD, 1, lane), acc);
      const int n = lane & 15, mb = (lane & 16) ? 8 : 0;
      const size_t outBase = ((size_t)b * cL + l0) * cE + (size_t)h * cD;
#pragma unroll
      for (int r = 0; r < 8; ++r) {
        const int li = mt + mb + r, ld = nt + n;
        attnb[outBase + (size_t)li * cE + ld] = (__bf16)(acc[r] / sDen[li]);
      }
    }
    __syncthreads();

    // S += k^T @ v   (A(m,k=j)=sK[j*cM+m]; B(k=j,n=d)=sV[j*cD+d])
    for (int t = wave; t < 64; t += 8) {
      const int mt = (t >> 2) << 4, nt = (t & 3) << 4;
      f32x8 acc = ldC(S + mt * cD + nt, cD, 1, lane);
#pragma unroll
      for (int kk = 0; kk < 64; kk += 32)
        acc = wmma_bf(frag_a_bf(sK + kk * cM + mt, 1, cM, lane),
                      frag_b_bf(sV + kk * cD + nt, cD, 1, lane), acc);
      stC(S + mt * cD + nt, cD, 1, lane, acc);
    }
    // z += column sums of k
    {
      float s = 0.f;
      for (int j = 0; j < 64; ++j) s += (float)sK[j * cM + tid];
      zv[tid] += s;
    }
    __syncthreads();
  }
}

// ---------------- kernel 4: out = attn @ Wout^T + bout ----------------
__global__ void __launch_bounds__(256)
k4_out(const __bf16* __restrict__ attnb, const float* __restrict__ Wout,
       const float* __restrict__ bout, float* __restrict__ out) {
  const int tid = threadIdx.x, wave = tid >> 5, lane = tid & 31;
  const int r0 = blockIdx.x * 64;   // row in [B*L]
  const int o0 = blockIdx.y * 64;   // output feature

  __shared__ __bf16 sA[64 * 64];
  __shared__ __bf16 sB[64 * 64];    // Wout tile, rows = out feature, cols = e

  f32x8 acc0 = zero8(), acc1 = zero8();
  for (int kt = 0; kt < cE; kt += 64) {
    for (int i = tid; i < 64 * 64; i += 256) {
      const int r = i >> 6, cc = i & 63;
      sA[i] = attnb[(size_t)(r0 + r) * cE + kt + cc];
      sB[i] = (__bf16)Wout[(size_t)(o0 + r) * cE + kt + cc];
    }
    if (kt + 64 < cE) {  // prefetch next K tiles (global_prefetch_b8)
      const int r = tid & 63;
      __builtin_prefetch(attnb + (size_t)(r0 + r) * cE + kt + 64, 0, 0);
      __builtin_prefetch(Wout + (size_t)(o0 + r) * cE + kt + 64, 0, 0);
    }
    __syncthreads();
#pragma unroll
    for (int half = 0; half < 2; ++half) {
      const int t = wave + half * 8;
      const int mt = (t >> 2) << 4, nt = (t & 3) << 4;
      f32x8 acc = half ? acc1 : acc0;
#pragma unroll
      for (int kk = 0; kk < 64; kk += 32)
        acc = wmma_bf(frag_a_bf(sA + mt * 64 + kk, 64, 1, lane),
                      frag_b_bf(sB + nt * 64 + kk, 1, 64, lane), acc);
      if (half) acc1 = acc; else acc0 = acc;
    }
    __syncthreads();
  }
#pragma unroll
  for (int half = 0; half < 2; ++half) {
    const int t = wave + half * 8;
    const int mt = (t >> 2) << 4, nt = (t & 3) << 4;
    const f32x8 acc = half ? acc1 : acc0;
    const int n = lane & 15, mb = (lane & 16) ? 8 : 0;
#pragma unroll
    for (int r = 0; r < 8; ++r)
      out[(size_t)(r0 + mt + mb + r) * cE + o0 + nt + n] = acc[r] + bout[o0 + nt + n];
  }
}

// ---------------- host launcher ----------------
extern "C" void kernel_launch(void* const* d_in, const int* in_sizes, int n_in,
                              void* d_out, int out_size, void* d_ws, size_t ws_size,
                              hipStream_t stream) {
  const float* qx   = (const float*)d_in[0];
  const float* kx   = (const float*)d_in[1];
  const float* vx   = (const float*)d_in[2];
  const float* Wq   = (const float*)d_in[3];
  const float* Wk   = (const float*)d_in[4];
  const float* Wv   = (const float*)d_in[5];
  const float* Wout = (const float*)d_in[6];
  const float* bout = (const float*)d_in[7];
  const float* proj = (const float*)d_in[8];

  char* w = (char*)d_ws;
  __bf16* qphi = (__bf16*)(w + OFF_QPHI);
  __bf16* kphi = (__bf16*)(w + OFF_KPHI);
  __bf16* vb   = (__bf16*)(w + OFF_VB);
  float*  khb  = (float*)(w + OFF_KH);
  float*  kd   = (float*)(w + OFF_KD);
  __bf16* attn = (__bf16*)(w + OFF_ATTN);
  int*    gmax = (int*)(w + OFF_GMAX);
  float*  Sc   = (float*)(w + OFF_SC);
  float*  zc   = (float*)(w + OFF_ZC);

  k0_init<<<dim3(1), dim3(1), 0, stream>>>(gmax);

  dim3 g1(cL / 64, cB * cH);
  k1_qkv_phi<<<g1, dim3(256), 0, stream>>>(qx, kx, vx, Wq, Wk, Wv, proj,
                                           qphi, khb, vb, kd, gmax);
  k2_kphi<<<g1, dim3(256), 0, stream>>>(khb, kd, proj, gmax, kphi);

  dim3 g3(cNCH, cB * cH);
  k3a_chunk<<<g3, dim3(256), 0, stream>>>(kphi, vb, Sc, zc);
  k3b_prefix<<<dim3(cB * cH), dim3(256), 0, stream>>>(Sc, zc);
  k3c_out<<<g3, dim3(256), 0, stream>>>(qphi, kphi, vb, Sc, zc, attn);

  k4_out<<<dim3((cB * cL) / 64, cE / 64), dim3(256), 0, stream>>>(attn, Wout, bout,
                                                                  (float*)d_out);
}
